// Neighbourhood_40561671143691
// MI455X (gfx1250) — compile-verified, matrix operand order checked by reading, not run on hardware
//
#include <hip/hip_runtime.h>
#include <hip/hip_bf16.h>

#define B_      4
#define N_      4096
#define C_      64
#define K_      64
#define TILE    16
#define CHUNK   512
#define NCHUNK  (N_ / CHUNK)
#define DSTRIDE (CHUNK + 2)      // pad: lanes L and L+16 (same col, rows +8) hit different banks
#define KEEP    32               // k / stride
#define SENTINEL 3.0e38f
#define PTS_ELEMS (CHUNK * 3)    // 1536 floats per staging buffer (6 KB)
#define TOT_ELEMS (N_ * 3)       // full per-batch tensor length in elements

#if defined(__has_builtin)
#if __has_builtin(__builtin_amdgcn_wmma_f32_16x16x4_f32)
#define USE_WMMA_F32X4 1
#endif
#if __has_builtin(__builtin_amdgcn_tensor_load_to_lds)
#define USE_TDM 1
#endif
#endif

typedef __attribute__((ext_vector_type(2)))  float    v2f;
typedef __attribute__((ext_vector_type(8)))  float    v8f;
typedef __attribute__((ext_vector_type(16))) _Float16 v16h;

#ifdef USE_TDM
typedef __attribute__((ext_vector_type(4))) unsigned int v4u;
typedef __attribute__((ext_vector_type(8))) int          v8i;
typedef __attribute__((ext_vector_type(4))) int          v4i;

// Issue one TDM descriptor: 1-D tile of PTS_ELEMS f32 from `gsrc` -> LDS byte
// address `lds_addr`. D# layout per CDNA5 ISA ch.8 (group0: count/lds/global/type,
// group1: data_size=4B, tensor_dim0, tile_dim0, dim0 stride; groups 2/3 unused).
__device__ __forceinline__ void tdm_load_points(const float* gsrc, unsigned int lds_addr) {
    const unsigned long long ga = (unsigned long long)(size_t)gsrc;
    v4u g0;
    g0[0] = 1u;                                                      // count=1 (user D#)
    g0[1] = lds_addr;                                                // LDS byte address
    g0[2] = (unsigned int)ga;                                        // global_addr[31:0]
    g0[3] = (unsigned int)((ga >> 32) & 0x01FFFFFFu) | (2u << 30);   // addr[56:32] | type=2
    v8i g1;
    g1[0] = (int)(2u << 16);                             // wg_mask=0, data_size=2 (4B)
    g1[1] = (int)((unsigned)(TOT_ELEMS & 0xFFFF) << 16); // tensor_dim0[15:0] @ [31:16]
    g1[2] = (int)((unsigned)(TOT_ELEMS >> 16) | (1u << 16)); // tensor_dim0 hi | tensor_dim1=1
    g1[3] = (int)((unsigned)PTS_ELEMS << 16);            // tensor_dim1 hi=0 | tile_dim0=1536
    g1[4] = 0;                                           // tile_dim1=0, tile_dim2=0 (1-D tile)
    g1[5] = (int)TOT_ELEMS;                              // tensor_dim0_stride lo32
    g1[6] = 0;                                           // stride hi | dim1_stride lo
    g1[7] = 0;
    v4i z4 = {};
#if defined(__clang_major__) && (__clang_major__ >= 23)
    v8i z8 = {};
    __builtin_amdgcn_tensor_load_to_lds(g0, g1, z4, z4, z8, 0);
#else
    __builtin_amdgcn_tensor_load_to_lds(g0, g1, z4, z4, 0);
#endif
}
#endif  // USE_TDM

// PERM2[j] = PERM[2*j], where PERM = jax.random.permutation(jax.random.key(1), 64)
// (host-precomputed fixed permutation; full PERM:
//  {37,5,52,18,9,60,27,44,1,33,14,58,23,41,7,50,29,12,63,35,2,47,20,55,10,38,25,0,
//   53,16,42,31,6,59,22,45,13,34,3,56,28,49,17,40,8,62,36,21,51,4,30,57,11,46,24,39,
//   15,61,32,48,19,43,26,54})
__device__ __constant__ int PERM2[KEEP] = {
    37, 52,  9, 27,  1, 14, 23,  7, 29, 63,  2, 20, 10, 25, 53, 42,
     6, 22, 13,  3, 28, 17,  8, 36, 51, 30, 11, 24, 15, 32, 19, 26};

__global__ __launch_bounds__(256) void knn_wmma_gather_kernel(
    const float* __restrict__ pts,   // [B, N, 3]
    const float* __restrict__ feat,  // [B, N, 64]
    float* __restrict__ out)         // [B, N, 32, 64]
{
    __shared__ float s_dist[TILE * DSTRIDE];  // d2 for 16 queries x 512 candidates
    __shared__ float s_topd[TILE * K_];       // running top-64 distances (sorted asc)
    __shared__ int   s_topi[TILE * K_];       // running top-64 indices
    __shared__ float s_sqm[TILE];             // |q_m|^2
#ifdef USE_TDM
    __shared__ float s_pts[2 * PTS_ELEMS];    // TDM-staged candidate points, ping/pong
#endif

    const int tid  = threadIdx.x;
    const int lane = tid & 31;
    const int wave = tid >> 5;
    const int hi   = lane >> 4;   // half-wave: selects K-pair {0,1} vs {2,3}
    const int l16  = lane & 15;

    const int b     = blockIdx.x >> 8;          // 256 query tiles per batch
    const int qBase = (blockIdx.x & 255) * TILE;

    for (int i = tid; i < TILE * K_; i += 256) {
        s_topd[i] = SENTINEL;
        s_topi[i] = 0x7fffffff;
    }
    if (tid < TILE) {
        size_t r = (size_t)(b * N_ + qBase + tid) * 3;
        float p0 = pts[r], p1 = pts[r + 1], p2 = pts[r + 2];
        s_sqm[tid] = p0 * p0 + p1 * p1 + p2 * p2;
    }
#ifdef USE_TDM
    if (tid < 32) {  // wave 0 issues the chunk-0 stage and owns its TENSORcnt
        tdm_load_points(pts + (size_t)(b * N_) * 3,
                        (unsigned int)(size_t)&s_pts[0]);
        __builtin_amdgcn_s_wait_tensorcnt(0);
    }
#endif
    __syncthreads();

    float smr[8];
#pragma unroll
    for (int r = 0; r < 8; ++r) smr[r] = s_sqm[r + 8 * hi];  // row m = r + 8*hi (C layout)

    // A tile (16 query rows x K=4, K=3 zero-padded). 32-bit A 16x4 layout:
    // lanes 0-15 hold (K0,K1), lanes 16-31 hold (K2,K3) for row M = lane%16.
    size_t rowA = (size_t)(b * N_ + qBase + l16) * 3;
    float ap0 = pts[rowA], ap1 = pts[rowA + 1], ap2 = pts[rowA + 2];
#ifdef USE_WMMA_F32X4
    v2f av;
    av.x = hi ? ap2 : ap0;
    av.y = hi ? 0.0f : ap1;
#else
    v16h a16 = {};
    if (!hi) { a16[0] = (_Float16)ap0; a16[1] = (_Float16)ap1; a16[2] = (_Float16)ap2; }
#endif

    for (int ch = 0; ch < NCHUNK; ++ch) {
        const int cb = ch * CHUNK;

#ifdef USE_TDM
        // Prefetch next chunk's points into the other buffer; hidden under this
        // chunk's WMMA + merge work. Target buffer was last read in chunk ch-1's
        // distance phase, protected by that chunk's closing barrier.
        if (tid < 32 && (ch + 1) < NCHUNK) {
            tdm_load_points(pts + (size_t)(b * N_ + cb + CHUNK) * 3,
                            (unsigned int)(size_t)&s_pts[((ch + 1) & 1) * PTS_ELEMS]);
        }
        const int pbase = (ch & 1) * PTS_ELEMS;
#endif

        // ---- distance tiles: one v_wmma per 16x16 tile, 4 tiles per wave ----
#pragma unroll
        for (int tt = 0; tt < 4; ++tt) {
            const int t  = wave * 4 + tt;
            const int nl = t * TILE + l16;   // chunk-local candidate column
            float q0, q1, q2;
#ifdef USE_TDM
            {
                const float* pp = &s_pts[pbase + nl * 3];
                q0 = pp[0]; q1 = pp[1]; q2 = pp[2];
            }
#else
            {
                size_t rowB = (size_t)(b * N_ + cb + nl) * 3;
                q0 = pts[rowB]; q1 = pts[rowB + 1]; q2 = pts[rowB + 2];
            }
#endif
            float sqn = q0 * q0 + q1 * q1 + q2 * q2;
            v8f acc = {};
#ifdef USE_WMMA_F32X4
            v2f bv;
            bv.x = hi ? q2 : q0;
            bv.y = hi ? 0.0f : q1;
            acc = __builtin_amdgcn_wmma_f32_16x16x4_f32(
                false, av, false, bv, (short)0, acc, false, false);
#else
            v16h b16 = {};
            if (!hi) { b16[0] = (_Float16)q0; b16[1] = (_Float16)q1; b16[2] = (_Float16)q2; }
            acc = __builtin_amdgcn_wmma_f32_16x16x32_f16(
                false, a16, false, b16, (short)0, acc, false, false);
#endif
            // C/D layout: VGPR r, lane L -> M = r + 8*(L/16), N = L%16
#pragma unroll
            for (int r = 0; r < 8; ++r) {
                float d2 = smr[r] + sqn - 2.0f * acc[r];
                s_dist[(r + 8 * hi) * DSTRIDE + nl] = d2;
            }
        }
        __syncthreads();

        // ---- streaming top-64 merge: each wave owns 2 queries ----
        for (int qq = 0; qq < 2; ++qq) {
            const int q = wave * 2 + qq;
            // 512 chunk values + 64 running-top values, 18 slots per lane
            float dl[18]; int il[18];
#pragma unroll
            for (int j = 0; j < 16; ++j) {
                int pos = lane + j * 32;
                dl[j] = s_dist[q * DSTRIDE + pos];
                il[j] = cb + pos;
            }
#pragma unroll
            for (int j = 16; j < 18; ++j) {
                int tp = lane + (j - 16) * 32;
                dl[j] = s_topd[q * K_ + tp];
                il[j] = s_topi[q * K_ + tp];
            }
            for (int k = 0; k < K_; ++k) {
                float bd = SENTINEL; int bi = 0x7fffffff; int bp = -1;
#pragma unroll
                for (int j = 0; j < 18; ++j) {
                    bool better = (dl[j] < bd) || (dl[j] == bd && il[j] < bi);
                    if (better) { bd = dl[j]; bi = il[j]; bp = lane + j * 32; }
                }
                // wave32 butterfly min-reduce (dist, then index tie-break)
#pragma unroll
                for (int off = 16; off > 0; off >>= 1) {
                    float od = __shfl_xor(bd, off, 32);
                    int   oi = __shfl_xor(bi, off, 32);
                    int   op = __shfl_xor(bp, off, 32);
                    bool better = (od < bd) || (od == bd && oi < bi);
                    if (better) { bd = od; bi = oi; bp = op; }
                }
                if (lane == 0) { s_topd[q * K_ + k] = bd; s_topi[q * K_ + k] = bi; }
                const bool own = (bp & 31) == lane;
                const int  js  = bp >> 5;
#pragma unroll
                for (int j = 0; j < 18; ++j)
                    if (own && js == j) dl[j] = SENTINEL;  // consume winner
            }
        }
#ifdef USE_TDM
        if (tid < 32) __builtin_amdgcn_s_wait_tensorcnt(0);  // next chunk staged
#endif
        __syncthreads();
    }

    // ---- permuted, strided, coalesced feature gather (reads hit L2: feat = 4 MB) ----
    const int c4 = tid & 15;  // float4 index within the 64 channels
    for (int p = tid >> 4; p < TILE * KEEP; p += 16) {
        const int q = p >> 5;
        const int j = p & 31;
        const int src = s_topi[q * K_ + PERM2[j]];
        const float4 v =
            ((const float4*)(feat + (size_t)(b * N_ + src) * C_))[c4];
        ((float4*)(out + ((size_t)(b * N_ + qBase + q) * KEEP + j) * C_))[c4] = v;
    }
}

extern "C" void kernel_launch(void* const* d_in, const int* in_sizes, int n_in,
                              void* d_out, int out_size, void* d_ws, size_t ws_size,
                              hipStream_t stream) {
    (void)in_sizes; (void)n_in; (void)out_size; (void)d_ws; (void)ws_size;
    const float* pts  = (const float*)d_in[0];  // [4,4096,3] f32
    const float* feat = (const float*)d_in[1];  // [4,4096,64] f32
    float* out = (float*)d_out;                 // [4,4096,32,64] f32
    // d_in[2] = k (64), d_in[3] = stride (2): fixed by the reference, hardcoded.
    knn_wmma_gather_kernel<<<dim3(B_ * (N_ / TILE)), dim3(256), 0, stream>>>(
        pts, feat, out);
}
